// AttentionPairBias_56736517980630
// MI455X (gfx1250) — compile-verified, matrix operand order checked by reading, not run on hardware
//
#include <hip/hip_runtime.h>
#include <math.h>

// ---------------------------------------------------------------------------
// CDNA5 (gfx1250) wave32 WMMA attention-with-pair-bias
// ---------------------------------------------------------------------------
#define Nseq 1024
#define Hh   16
#define HD   32
#define Dd   512
#define DZ   128
#define ZPAD 132   // 128 dwords + 4 dword TDM pad per row -> conflict-free LDS banks

typedef __attribute__((ext_vector_type(16))) __bf16 v16bf;
typedef __attribute__((ext_vector_type(8)))  float  v8f;
typedef __attribute__((ext_vector_type(8)))  unsigned short us8;
typedef __attribute__((ext_vector_type(4)))  unsigned int u32x4;
typedef __attribute__((ext_vector_type(8)))  int i32x8;
typedef __attribute__((ext_vector_type(4)))  int i32x4;

#ifndef USE_TDM
#if defined(__has_builtin)
#if __has_builtin(__builtin_amdgcn_tensor_load_to_lds)
#define USE_TDM 1
#endif
#endif
#endif
#ifndef USE_TDM
#define USE_TDM 0
#endif

union FragU {
    v16bf v;
    us8 h[2];
    unsigned short u[16];
};

static __device__ __forceinline__ unsigned short f2bf(float f) {
    unsigned int u = __float_as_uint(f);
    unsigned int r = (u + 0x7FFFu + ((u >> 16) & 1u)) >> 16;   // RNE
    return (unsigned short)r;
}

// Build a 16x32 bf16 A/B fragment chunk from a row-major bf16 row.
// Lane layout (ISA 7.12.2): lanes 0-15 hold K = kb+[0..7], kb+16+[0..7];
// lanes 16-31 hold K = kb+8+[0..7], kb+24+[0..7]  -> offset = kb + half*8 (+16)
static __device__ __forceinline__ v16bf frag_load(const unsigned short* row, int kb, int half) {
    FragU f;
    f.h[0] = *reinterpret_cast<const us8*>(row + kb + half * 8);
    f.h[1] = *reinterpret_cast<const us8*>(row + kb + 16 + half * 8);
    return f.v;
}

// ---------------------------------------------------------------------------
// LayerNorm over s rows (D=512) -> bf16
// ---------------------------------------------------------------------------
__global__ void k_ln_s(const float* __restrict__ s, const float* __restrict__ w,
                       const float* __restrict__ b, unsigned short* __restrict__ out) {
    const int n = blockIdx.x;
    const int t = threadIdx.x;                       // 128 threads
    const float* row = s + (size_t)n * Dd;
    float x[4], s1 = 0.f, s2 = 0.f;
#pragma unroll
    for (int i = 0; i < 4; ++i) {
        x[i] = row[t + i * 128];
        s1 += x[i];
        s2 += x[i] * x[i];
    }
    __shared__ float r1[128], r2[128];
    __shared__ float mv[2];
    r1[t] = s1; r2[t] = s2;
    __syncthreads();
    for (int off = 64; off > 0; off >>= 1) {
        if (t < off) { r1[t] += r1[t + off]; r2[t] += r2[t + off]; }
        __syncthreads();
    }
    if (t == 0) {
        float mean = r1[0] * (1.f / Dd);
        float var  = r2[0] * (1.f / Dd) - mean * mean;
        mv[0] = mean; mv[1] = rsqrtf(var + 1e-5f);
    }
    __syncthreads();
    const float mean = mv[0], rstd = mv[1];
#pragma unroll
    for (int i = 0; i < 4; ++i) {
        const int c = t + i * 128;
        out[(size_t)n * Dd + c] = f2bf((x[i] - mean) * rstd * w[c] + b[c]);
    }
}

// ---------------------------------------------------------------------------
// Transpose + f32->bf16:  out[r][c] = bf16(in[c][r]),  out is R x C
// ---------------------------------------------------------------------------
__global__ void k_tr_cvt(const float* __restrict__ in, unsigned short* __restrict__ out,
                         int R, int C) {
    long long idx = (long long)blockIdx.x * 256 + threadIdx.x;
    if (idx >= (long long)R * C) return;
    int r = (int)(idx / C);
    int c = (int)(idx - (long long)r * C);
    out[idx] = f2bf(in[(long long)c * R + r]);
}

// ---------------------------------------------------------------------------
// Generic wave32 WMMA bf16 GEMM:  C[M][Nc] = A[M][K] * Bt[Nc][K]^T
// One 16x16 output tile per wave; 8 waves/block. Optional bias, f32 or bf16 out.
// ---------------------------------------------------------------------------
__global__ void k_wmma_gemm(const unsigned short* __restrict__ A,
                            const unsigned short* __restrict__ Bt,
                            float* __restrict__ Cf, unsigned short* __restrict__ Cb,
                            const float* __restrict__ bias,
                            int M, int Nc, int Kd,
                            long long sA, long long sB, long long sC) {
    const int lane = threadIdx.x & 31;
    const int wave = threadIdx.x >> 5;
    const int half = lane >> 4;
    const int lr = lane & 15;
    const int ntn = Nc >> 4;
    const int tiles = (M >> 4) * ntn;
    const int id = blockIdx.x * (blockDim.x >> 5) + wave;
    if (id >= tiles) return;
    const int tm = id / ntn;
    const int tn = id - tm * ntn;
    const long long zb = blockIdx.z;

    const unsigned short* Ar = A + zb * sA + (long long)(tm * 16 + lr) * Kd;
    const unsigned short* Br = Bt + zb * sB + (long long)(tn * 16 + lr) * Kd;

    v8f acc = {};
    for (int kb = 0; kb < Kd; kb += 32) {
        v16bf a = frag_load(Ar, kb, half);
        v16bf b = frag_load(Br, kb, half);
        acc = __builtin_amdgcn_wmma_f32_16x16x32_bf16(false, a, false, b,
                                                      (short)0, acc, false, false);
    }
    const int col = tn * 16 + lr;
    const float bv = bias ? bias[col] : 0.f;
#pragma unroll
    for (int r = 0; r < 8; ++r) {
        const int row = tm * 16 + r + 8 * half;      // C layout: VGPR r -> M=r (+8 upper half)
        const long long o = zb * sC + (long long)row * Nc + col;
        const float v = acc[r] + bv;
        if (Cf) Cf[o] = v;
        else    Cb[o] = f2bf(v);
    }
}

// ---------------------------------------------------------------------------
// Pair bias: TDM-stage a 16x128 f32 z-tile to LDS (padded pitch 132 dwords),
// per-(i,j) LayerNorm in registers, WMMA vs WzT[16][128], coalesced S stores.
// 4 waves / 128 threads per block; one 16-pair tile per wave.
// ---------------------------------------------------------------------------
__global__ void __launch_bounds__(128) k_zbias(
        const float* __restrict__ z, const float* __restrict__ zw,
        const float* __restrict__ zbv, const unsigned short* __restrict__ WzT,
        float* __restrict__ S) {
    const int lane = threadIdx.x & 31;
    const int wave = threadIdx.x >> 5;
    const int half = lane >> 4;
    const int lr = lane & 15;
    const int i = blockIdx.y;
    const int jt = blockIdx.x * 4 + wave;
    const int j0 = jt * 16;

    __shared__ float tr[4][16][17];                  // output transpose buffer

#if USE_TDM
    __shared__ float tile[4][16 * ZPAD];             // 16 rows x (128+4) dwords per wave
    {
        // ---- Tensor DMA descriptor (ISA 8.3-8.5): 2D tile 128x16 f32 ----
        const unsigned long long ga =
            (unsigned long long)(const void*)(z + ((size_t)i * Nseq + j0) * DZ);
        const unsigned lds_off = (unsigned)(size_t)(void*)&tile[wave][0]; // LDS aperture low bits

        u32x4 g0;
        g0[0] = 1u;                                          // count=1, user descriptor
        g0[1] = lds_off;                                     // lds_addr
        g0[2] = (unsigned)(ga & 0xFFFFFFFFu);                // global_addr[31:0]
        g0[3] = (unsigned)((ga >> 32) & 0x01FFFFFFu)         // global_addr[56:32]
              | (2u << 30);                                  // type=2 (image/TDM)

        i32x8 g1;
        g1[0] = (int)((2u << 16)                             // data_size = 4B
              | (1u << 20)                                   // pad_enable
              | (6u << 22)                                   // pad_interval: 128 dwords
              | (3u << 25));                                 // pad_amount: 4 dwords
        g1[1] = (int)((unsigned)DZ << 16);                   // tensor_dim0 = 128 (lo16)
        g1[2] = 0;                                           // dim0 hi | dim1 lo16 (=0)
        g1[3] = (int)((0x10u) | ((unsigned)DZ << 16));       // dim1 hi16 (1M rows) | tile_dim0=128
        g1[4] = 16;                                          // tile_dim1 = 16, tile_dim2 = 0
        g1[5] = DZ;                                          // tensor_dim0_stride = 128
        g1[6] = 0;
        g1[7] = 0;

        i32x4 g2 = {0, 0, 0, 0};
        i32x4 g3 = {0, 0, 0, 0};
        i32x8 g4 = {0, 0, 0, 0, 0, 0, 0, 0};                 // unused ext group (6-arg form)
        __builtin_amdgcn_tensor_load_to_lds(g0, g1, g2, g3, g4, 0);
        __builtin_amdgcn_s_wait_tensorcnt(0);
    }
    __syncthreads();
    const float* src = &tile[wave][lr * ZPAD];       // row j0+lr, conflict-free pitch
#else
    const float* src = z + ((size_t)i * Nseq + (j0 + lr)) * DZ;
    __builtin_prefetch(src + 16 * DZ, 0, 1);
#endif

    float x[64], s1 = 0.f, s2 = 0.f;
#pragma unroll
    for (int s = 0; s < 4; ++s) {
        const int b0 = s * 32 + half * 8;
        const int b1 = s * 32 + 16 + half * 8;
#pragma unroll
        for (int t = 0; t < 8; ++t) {
#if USE_TDM
            const float a0 = src[b0 + t];
            const float a1 = src[b1 + t];
#else
            const float a0 = __builtin_nontemporal_load(&src[b0 + t]);  // z streamed once
            const float a1 = __builtin_nontemporal_load(&src[b1 + t]);
#endif
            x[s * 16 + t] = a0;
            x[s * 16 + 8 + t] = a1;
            s1 += a0 + a1;
            s2 += a0 * a0 + a1 * a1;
        }
    }
    // lanes L and L+16 hold complementary halves of the same row j0+lr
    s1 += __shfl_xor(s1, 16, 32);
    s2 += __shfl_xor(s2, 16, 32);
    const float mean = s1 * (1.f / DZ);
    const float var  = s2 * (1.f / DZ) - mean * mean;
    const float rstd = rsqrtf(var + 1e-5f);

    v8f acc = {};
    const unsigned short* wrow = WzT + lr * DZ;       // head lr's column of Wz
#pragma unroll
    for (int s = 0; s < 4; ++s) {
        FragU fa;
        const int b0 = s * 32 + half * 8;
        const int b1 = s * 32 + 16 + half * 8;
#pragma unroll
        for (int t = 0; t < 8; ++t) {
            fa.u[t]     = f2bf((x[s * 16 + t]     - mean) * rstd * zw[b0 + t] + zbv[b0 + t]);
            fa.u[8 + t] = f2bf((x[s * 16 + 8 + t] - mean) * rstd * zw[b1 + t] + zbv[b1 + t]);
        }
        v16bf fb = frag_load(wrow, s * 32, half);
        acc = __builtin_amdgcn_wmma_f32_16x16x32_bf16(false, fa.v, false, fb,
                                                      (short)0, acc, false, false);
    }

    // Transpose 16(j) x 16(head) tile through LDS -> coalesced per-head stores
#pragma unroll
    for (int r = 0; r < 8; ++r)
        tr[wave][lr][r + 8 * half] = acc[r];          // [head][j_local]
    __syncthreads();
    const size_t NN = (size_t)Nseq * Nseq;
#pragma unroll
    for (int r = 0; r < 8; ++r) {
        const int head = r + 8 * half;
        S[(size_t)head * NN + (size_t)i * Nseq + j0 + lr] = tr[wave][head][lr];
    }
}

// ---------------------------------------------------------------------------
// Scores: S[h][q][k] += (q . k) / sqrt(HD).  K-dim = HD = 32 -> single WMMA.
// ---------------------------------------------------------------------------
__global__ void k_scores(const unsigned short* __restrict__ qb,
                         const unsigned short* __restrict__ kb,
                         float* __restrict__ S) {
    const int lane = threadIdx.x & 31;
    const int wave = threadIdx.x >> 5;
    const int half = lane >> 4;
    const int lr = lane & 15;
    const int h = blockIdx.z;
    const int qt = blockIdx.y;
    const int kt = blockIdx.x * 8 + wave;

    const unsigned short* qr = qb + (size_t)(qt * 16 + lr) * Dd + h * HD;
    const unsigned short* kr = kb + (size_t)(kt * 16 + lr) * Dd + h * HD;
    v16bf a = frag_load(qr, 0, half);
    v16bf b = frag_load(kr, 0, half);
    v8f acc = {};
    acc = __builtin_amdgcn_wmma_f32_16x16x32_bf16(false, a, false, b,
                                                  (short)0, acc, false, false);
    const float scale = 0.17677669529663687f;         // HD^-0.5
    const size_t NN = (size_t)Nseq * Nseq;
    const size_t base = (size_t)h * NN;
    const int col = kt * 16 + lr;
#pragma unroll
    for (int r = 0; r < 8; ++r) {
        const int row = qt * 16 + r + 8 * half;
        const size_t o = base + (size_t)row * Nseq + col;
        S[o] = S[o] + acc[r] * scale;                 // pair bias already in S
    }
}

// ---------------------------------------------------------------------------
// Row softmax over k (length N), f32 in, bf16 out
// ---------------------------------------------------------------------------
__global__ void k_softmax(const float* __restrict__ S, unsigned short* __restrict__ P) {
    const int row = blockIdx.x;                       // 0..H*N-1
    const int t = threadIdx.x;                        // 256
    const float* src = S + (size_t)row * Nseq;
    float x[4], vmax = -3.4e38f;
#pragma unroll
    for (int i = 0; i < 4; ++i) {
        x[i] = src[t + i * 256];
        vmax = fmaxf(vmax, x[i]);
    }
    __shared__ float red[256];
    __shared__ float bc[2];
    red[t] = vmax;
    __syncthreads();
    for (int off = 128; off > 0; off >>= 1) {
        if (t < off) red[t] = fmaxf(red[t], red[t + off]);
        __syncthreads();
    }
    if (t == 0) bc[0] = red[0];
    __syncthreads();
    vmax = bc[0];
    float e[4], lsum = 0.f;
#pragma unroll
    for (int i = 0; i < 4; ++i) {
        e[i] = __expf(x[i] - vmax);
        lsum += e[i];
    }
    __syncthreads();
    red[t] = lsum;
    __syncthreads();
    for (int off = 128; off > 0; off >>= 1) {
        if (t < off) red[t] += red[t + off];
        __syncthreads();
    }
    if (t == 0) bc[1] = 1.f / red[0];
    __syncthreads();
    const float inv = bc[1];
#pragma unroll
    for (int i = 0; i < 4; ++i)
        P[(size_t)row * Nseq + t + i * 256] = f2bf(e[i] * inv);
}

// ---------------------------------------------------------------------------
// Gate + pack: gated[n][c] = bf16(o[h][n][d] * sigmoid(g[n][c])),  c = h*32+d
// ---------------------------------------------------------------------------
__global__ void k_gate(const float* __restrict__ o, const float* __restrict__ g,
                       unsigned short* __restrict__ out) {
    const long long idx = (long long)blockIdx.x * 256 + threadIdx.x;  // N*D
    const int n = (int)(idx >> 9);
    const int c = (int)(idx & 511);
    const int h = c >> 5;
    const int d = c & 31;
    const float ov = o[((size_t)h * Nseq + n) * HD + d];
    const float gv = g[idx];
    const float sg = 1.f / (1.f + __expf(-gv));
    out[idx] = f2bf(ov * sg);
}

// ---------------------------------------------------------------------------
extern "C" void kernel_launch(void* const* d_in, const int* in_sizes, int n_in,
                              void* d_out, int out_size, void* d_ws, size_t ws_size,
                              hipStream_t stream) {
    (void)in_sizes; (void)n_in; (void)out_size; (void)ws_size;
    const float* s      = (const float*)d_in[0];
    const float* z      = (const float*)d_in[1];
    const float* nsw    = (const float*)d_in[2];
    const float* nsb    = (const float*)d_in[3];
    const float* Wq     = (const float*)d_in[4];
    const float* bq     = (const float*)d_in[5];
    const float* Wk     = (const float*)d_in[6];
    const float* Wv     = (const float*)d_in[7];
    const float* Wg     = (const float*)d_in[8];
    const float* znw    = (const float*)d_in[9];
    const float* znb    = (const float*)d_in[10];
    const float* Wz     = (const float*)d_in[11];
    const float* Wo     = (const float*)d_in[12];
    float* out          = (float*)d_out;

    char* base = (char*)d_ws;
    size_t off = 0;
    auto alloc = [&](size_t bytes) -> void* {
        off = (off + 255) & ~(size_t)255;
        void* p = base + off;
        off += bytes;
        return p;
    };
    unsigned short* sln   = (unsigned short*)alloc((size_t)Nseq * Dd * 2);
    unsigned short* WqT   = (unsigned short*)alloc((size_t)Dd * Dd * 2);
    unsigned short* WkT   = (unsigned short*)alloc((size_t)Dd * Dd * 2);
    unsigned short* WvT   = (unsigned short*)alloc((size_t)Dd * Dd * 2);
    unsigned short* WgT   = (unsigned short*)alloc((size_t)Dd * Dd * 2);
    unsigned short* WoT   = (unsigned short*)alloc((size_t)Dd * Dd * 2);
    unsigned short* WzT   = (unsigned short*)alloc((size_t)Hh * DZ * 2);
    unsigned short* qb    = (unsigned short*)alloc((size_t)Nseq * Dd * 2);
    unsigned short* kbuf  = (unsigned short*)alloc((size_t)Nseq * Dd * 2);
    float*          vf    = (float*)alloc((size_t)Nseq * Dd * 4);
    float*          gf    = (float*)alloc((size_t)Nseq * Dd * 4);
    unsigned short* vt    = (unsigned short*)alloc((size_t)Dd * Nseq * 2);
    float*          S     = (float*)alloc((size_t)Hh * Nseq * Nseq * 4);
    unsigned short* P     = (unsigned short*)alloc((size_t)Hh * Nseq * Nseq * 2);
    float*          of    = (float*)alloc((size_t)Hh * Nseq * HD * 4);
    unsigned short* gated = (unsigned short*)alloc((size_t)Nseq * Dd * 2);

    // 1. LayerNorm(s) -> bf16
    k_ln_s<<<Nseq, 128, 0, stream>>>(s, nsw, nsb, sln);

    // 2. Weight transposes to Bt layout (bf16)
    k_tr_cvt<<<1024, 256, 0, stream>>>(Wq, WqT, Dd, Dd);
    k_tr_cvt<<<1024, 256, 0, stream>>>(Wk, WkT, Dd, Dd);
    k_tr_cvt<<<1024, 256, 0, stream>>>(Wv, WvT, Dd, Dd);
    k_tr_cvt<<<1024, 256, 0, stream>>>(Wg, WgT, Dd, Dd);
    k_tr_cvt<<<1024, 256, 0, stream>>>(Wo, WoT, Dd, Dd);
    k_tr_cvt<<<8, 256, 0, stream>>>(Wz, WzT, Hh, DZ);

    // 3. Projections (M=1024, N=512, K=512): 2048 tiles / 8 waves -> 256 blocks
    k_wmma_gemm<<<dim3(256, 1, 1), 256, 0, stream>>>(sln, WqT, nullptr, qb,   bq,
                                                     Nseq, Dd, Dd, 0, 0, 0);
    k_wmma_gemm<<<dim3(256, 1, 1), 256, 0, stream>>>(sln, WkT, nullptr, kbuf, nullptr,
                                                     Nseq, Dd, Dd, 0, 0, 0);
    k_wmma_gemm<<<dim3(256, 1, 1), 256, 0, stream>>>(sln, WvT, vf, nullptr, nullptr,
                                                     Nseq, Dd, Dd, 0, 0, 0);
    k_wmma_gemm<<<dim3(256, 1, 1), 256, 0, stream>>>(sln, WgT, gf, nullptr, nullptr,
                                                     Nseq, Dd, Dd, 0, 0, 0);

    // 4. V -> transposed bf16 [D][N] (= per-head [HD][N] Bt for the PV GEMM)
    k_tr_cvt<<<2048, 256, 0, stream>>>(vf, vt, Dd, Nseq);

    // 5. Pair bias (dominant 512MB z read, TDM-staged) -> S[h][i][j]
    k_zbias<<<dim3(16, Nseq, 1), 128, 0, stream>>>(z, znw, znb, WzT, S);

    // 6. S += QK^T * scale
    k_scores<<<dim3(8, 64, Hh), 256, 0, stream>>>(qb, kbuf, S);

    // 7. softmax -> P (bf16)
    k_softmax<<<Hh * Nseq, 256, 0, stream>>>(S, P);

    // 8. O = P @ V per head: M=1024, N=32, K=1024 -> 128 tiles -> grid(16,1,16)
    k_wmma_gemm<<<dim3(16, 1, Hh), 256, 0, stream>>>(
        P, vt, of, nullptr, nullptr, Nseq, HD, Nseq,
        (long long)Nseq * Nseq, (long long)HD * Nseq, (long long)Nseq * HD);

    // 9. gate + pack to bf16
    k_gate<<<2048, 256, 0, stream>>>(of, gf, gated);

    // 10. out = gated @ Wo (f32 out)
    k_wmma_gemm<<<dim3(256, 1, 1), 256, 0, stream>>>(gated, WoT, out, nullptr, nullptr,
                                                     Nseq, Dd, Dd, 0, 0, 0);
}
